// ReversedAttention_85882166050866
// MI455X (gfx1250) — compile-verified, matrix operand order checked by reading, not run on hardware
//
#include <hip/hip_runtime.h>

// ---------------------------------------------------------------------------
// ReversedAttention for MI455X (gfx1250, wave32, WMMA f32_16x16x32_f16)
//   B=8, N=8192, D=256
//   q = x@Wq^T+bq ; k masked ; aw = softmax_d(K^T V / sqrt(B)) ; out = Q aw
//
// All GEMMs: 2x2 register blocking (wave owns 32x32 output), double-buffered
// LDS staging (fetch next K-tile's global loads before the WMMAs, commit the
// f16 conversion + ds_store after them).  Kernel 1 emits k/v already
// TRANSPOSED (kT[b][d][n], vT[b][e][n]) so the K^T V kernel stages with the
// same vectorized b128 path as everything else (no scalar LDS scatter).
// ---------------------------------------------------------------------------

#define BB 8
#define NN 8192
#define DD 256
#define LDS_STRIDE 40          // 32 K-values + pad (f16 units)
#define WPB 4                  // waves per block for GEMM kernels (128 thr)
#define SWPB 8                 // waves per block for softmax (256 thr)
#define KSPLIT 4               // split-K factor for the K^T V gram kernel

typedef __attribute__((ext_vector_type(16))) _Float16 v16h;
typedef __attribute__((ext_vector_type(8)))  _Float16 v8h;
typedef __attribute__((ext_vector_type(8)))  float    v8f;
typedef unsigned long long ull;

__device__ __forceinline__ void lds_fence() {
    asm volatile("s_wait_dscnt 0" ::: "memory");
}

// ---- fragment load: tile is T[16][LDS_STRIDE] (row = M for A / N-col for B,
// ---- inner dim = K).  Lane layout per CDNA5 ISA 7.12.2 (16-bit A/B):
// ----   row/col = lane&15 ; elem i<8 -> K = 8*(lane>>4)+i ; i>=8 -> K+16
__device__ __forceinline__ v16h load_frag(const _Float16* tile, int lane) {
    const int r = lane & 15, h = lane >> 4;
    const _Float16* p = tile + r * LDS_STRIDE + h * 8;
    v16h f;
#pragma unroll
    for (int i = 0; i < 8; ++i) { f[i] = p[i]; f[i + 8] = p[i + 16]; }
    return f;
}

// ---- split fetch/commit staging: 16 rows x 32 K tiles --------------------
struct F32Tile { float4 a, b, c, d; };

__device__ __forceinline__ F32Tile fetch_f32(const float* g, size_t ld, int lane) {
    const int r = lane & 15, h = lane >> 4;
    const float* s = g + (size_t)r * ld + h * 16;
    F32Tile t;
    t.a = *(const float4*)(s + 0);
    t.b = *(const float4*)(s + 4);
    t.c = *(const float4*)(s + 8);
    t.d = *(const float4*)(s + 12);
    return t;
}

__device__ __forceinline__ void commit_f32(_Float16* tile, const F32Tile& t, int lane) {
    const int r = lane & 15, h = lane >> 4;
    v8h lo, hi;
    lo[0] = (_Float16)t.a.x; lo[1] = (_Float16)t.a.y; lo[2] = (_Float16)t.a.z; lo[3] = (_Float16)t.a.w;
    lo[4] = (_Float16)t.b.x; lo[5] = (_Float16)t.b.y; lo[6] = (_Float16)t.b.z; lo[7] = (_Float16)t.b.w;
    hi[0] = (_Float16)t.c.x; hi[1] = (_Float16)t.c.y; hi[2] = (_Float16)t.c.z; hi[3] = (_Float16)t.c.w;
    hi[4] = (_Float16)t.d.x; hi[5] = (_Float16)t.d.y; hi[6] = (_Float16)t.d.z; hi[7] = (_Float16)t.d.w;
    v8h* p = (v8h*)(tile + r * LDS_STRIDE + h * 16);
    p[0] = lo;
    p[1] = hi;
}

struct F16Tile { v8h x0, x1; };

__device__ __forceinline__ F16Tile fetch_f16(const _Float16* g, size_t ld, int lane) {
    const int r = lane & 15, h = lane >> 4;
    const v8h* s = (const v8h*)(g + (size_t)r * ld + h * 16);
    F16Tile t;
    t.x0 = s[0];
    t.x1 = s[1];
    return t;
}

__device__ __forceinline__ void commit_f16(_Float16* tile, const F16Tile& t, int lane) {
    const int r = lane & 15, h = lane >> 4;
    v8h* p = (v8h*)(tile + r * LDS_STRIDE + h * 16);
    p[0] = t.x0;
    p[1] = t.x1;
}

__device__ __forceinline__ v8f wmma16(v16h a, v16h b, v8f c) {
    return __builtin_amdgcn_wmma_f32_16x16x32_f16(false, a, false, b,
                                                  (short)0, c, false, false);
}

// ===========================================================================
// Kernel 1: fused q/k/v projections.  y[r,e] = sum_d x[r,d]*W[e,d] + b[e]
//   q  -> row-major f16  qf[b*N + n][D]        (A-side of kernel 4)
//   k  -> TRANSPOSED f16 kT[b][d][n] (masked)  (A-side of kernel 2)
//   v  -> TRANSPOSED f16 vT[b][e][n]           (B-side of kernel 2)
// ===========================================================================
__global__ __launch_bounds__(128) void ra_proj_kernel(
    const float* __restrict__ query, const float* __restrict__ key,
    const float* __restrict__ value,
    const float* __restrict__ Wq, const float* __restrict__ bq,
    const float* __restrict__ Wk, const float* __restrict__ bk,
    const float* __restrict__ Wv, const float* __restrict__ bv,
    const unsigned char* __restrict__ mask,
    _Float16* __restrict__ qf, _Float16* __restrict__ kT,
    _Float16* __restrict__ vT) {
    __shared__ __align__(16) _Float16 lds[WPB][2][4][16 * LDS_STRIDE];
    const int lane = threadIdx.x & 31;
    const int wave = threadIdx.x >> 5;

    const int MB = (BB * NN) / 32;              // 2048 row-blocks
    const int NB = DD / 32;                     // 8 col-blocks
    int gt = blockIdx.x * WPB + wave;           // 0 .. 3*MB*NB-1 (exact)
    const int which = gt / (MB * NB);
    int rem = gt % (MB * NB);
    const int mb = rem / NB;
    const int nb = rem % NB;

    const float* X  = (which == 0) ? query : (which == 1) ? key : value;
    const float* W  = (which == 0) ? Wq : (which == 1) ? Wk : Wv;
    const float* BI = (which == 0) ? bq : (which == 1) ? bk : bv;

    const int row0 = mb * 32;                   // flat row over B*N
    const int col0 = nb * 32;
    const float bias0 = BI[col0 + (lane & 15)];
    const float bias1 = BI[col0 + 16 + (lane & 15)];

    const float* XA0 = X + (size_t)row0 * DD;
    const float* XA1 = X + (size_t)(row0 + 16) * DD;
    const float* WB0 = W + (size_t)col0 * DD;
    const float* WB1 = W + (size_t)(col0 + 16) * DD;

    v8f c00 = {}, c01 = {}, c10 = {}, c11 = {};
#pragma unroll
    for (int i = 0; i < 8; ++i) {
        c00[i] = bias0; c10[i] = bias0;
        c01[i] = bias1; c11[i] = bias1;
    }

    // prologue: stage K-step 0 into buffer 0
    F32Tile ra0 = fetch_f32(XA0, DD, lane), ra1 = fetch_f32(XA1, DD, lane);
    F32Tile rb0 = fetch_f32(WB0, DD, lane), rb1 = fetch_f32(WB1, DD, lane);
    commit_f32(lds[wave][0][0], ra0, lane);
    commit_f32(lds[wave][0][1], ra1, lane);
    commit_f32(lds[wave][0][2], rb0, lane);
    commit_f32(lds[wave][0][3], rb1, lane);

    int cur = 0;
#pragma unroll
    for (int kk = 0; kk < DD; kk += 32) {
        lds_fence();
        v16h a0 = load_frag(lds[wave][cur][0], lane);
        v16h a1 = load_frag(lds[wave][cur][1], lane);
        v16h b0 = load_frag(lds[wave][cur][2], lane);
        v16h b1 = load_frag(lds[wave][cur][3], lane);
        const bool nxt = (kk + 32) < DD;
        if (nxt) {                               // issue next global loads early
            ra0 = fetch_f32(XA0 + kk + 32, DD, lane);
            ra1 = fetch_f32(XA1 + kk + 32, DD, lane);
            rb0 = fetch_f32(WB0 + kk + 32, DD, lane);
            rb1 = fetch_f32(WB1 + kk + 32, DD, lane);
        }
        c00 = wmma16(a0, b0, c00);
        c01 = wmma16(a0, b1, c01);
        c10 = wmma16(a1, b0, c10);
        c11 = wmma16(a1, b1, c11);
        if (nxt) {                               // convert + stage after WMMAs
            commit_f32(lds[wave][cur ^ 1][0], ra0, lane);
            commit_f32(lds[wave][cur ^ 1][1], ra1, lane);
            commit_f32(lds[wave][cur ^ 1][2], rb0, lane);
            commit_f32(lds[wave][cur ^ 1][3], rb1, lane);
        }
        cur ^= 1;
    }

    const int e0 = col0 + (lane & 15);
    const int rb_lo = row0 + (lane >> 4) * 8;    // 8 consecutive flat rows
    if (which == 0) {
        // q: row-major [r][e]
#pragma unroll
        for (int v = 0; v < 8; ++v) {
            qf[(size_t)(rb_lo + v) * DD + e0]           = (_Float16)c00[v];
            qf[(size_t)(rb_lo + v) * DD + e0 + 16]      = (_Float16)c01[v];
            qf[(size_t)(rb_lo + 16 + v) * DD + e0]      = (_Float16)c10[v];
            qf[(size_t)(rb_lo + 16 + v) * DD + e0 + 16] = (_Float16)c11[v];
        }
    } else {
        // k/v: transposed [b][col][n]; each accumulator = 8 consecutive n
        ull m0 = *(const ull*)(mask + rb_lo);
        ull m1 = *(const ull*)(mask + rb_lo + 16);
        if (which == 2) { m0 = 0ull; m1 = 0ull; }
        v8h h00, h01, h10, h11;
#pragma unroll
        for (int v = 0; v < 8; ++v) {
            const bool z0 = (m0 >> (8 * v)) & 0xffull;
            const bool z1 = (m1 >> (8 * v)) & 0xffull;
            h00[v] = z0 ? (_Float16)0.0f : (_Float16)c00[v];
            h01[v] = z0 ? (_Float16)0.0f : (_Float16)c01[v];
            h10[v] = z1 ? (_Float16)0.0f : (_Float16)c10[v];
            h11[v] = z1 ? (_Float16)0.0f : (_Float16)c11[v];
        }
        const int b = row0 >> 13;                // row0 / NN
        const int n0 = (row0 & (NN - 1)) + (lane >> 4) * 8;
        _Float16* OT = ((which == 1) ? kT : vT) + (size_t)b * DD * NN;
        *(v8h*)(OT + (size_t)e0 * NN + n0)             = h00;
        *(v8h*)(OT + (size_t)(e0 + 16) * NN + n0)      = h01;
        *(v8h*)(OT + (size_t)e0 * NN + n0 + 16)        = h10;
        *(v8h*)(OT + (size_t)(e0 + 16) * NN + n0 + 16) = h11;
    }
}

// ===========================================================================
// Kernel 2: partial aw[ks][b,d,e] = sum_{n in slice} kT[b,d,n] * vT[b,e,n]
//   Row-major staging from the transposed layouts; K = n, 2048 per slice.
// ===========================================================================
__global__ __launch_bounds__(128) void ra_gram_kernel(
    const _Float16* __restrict__ kT, const _Float16* __restrict__ vT,
    float* __restrict__ awp) {
    __shared__ __align__(16) _Float16 lds[WPB][2][4][16 * LDS_STRIDE];
    const int lane = threadIdx.x & 31;
    const int wave = threadIdx.x >> 5;

    // gt = ((b*8 + db)*8 + eb)*KSPLIT + ks : 2048 waves
    int gt = blockIdx.x * WPB + wave;
    const int ks = gt & (KSPLIT - 1);
    int rem = gt / KSPLIT;
    const int eb = rem & 7; rem >>= 3;
    const int db = rem & 7; rem >>= 3;
    const int b  = rem;

    const int d0 = db * 32, e0 = eb * 32;
    const _Float16* kA0 = kT + ((size_t)b * DD + d0) * NN;
    const _Float16* kA1 = kT + ((size_t)b * DD + d0 + 16) * NN;
    const _Float16* vB0 = vT + ((size_t)b * DD + e0) * NN;
    const _Float16* vB1 = vT + ((size_t)b * DD + e0 + 16) * NN;

    v8f c00 = {}, c01 = {}, c10 = {}, c11 = {};
    const int nbeg = ks * (NN / KSPLIT);
    const int nend = nbeg + (NN / KSPLIT);

    F16Tile ra0 = fetch_f16(kA0 + nbeg, NN, lane), ra1 = fetch_f16(kA1 + nbeg, NN, lane);
    F16Tile rb0 = fetch_f16(vB0 + nbeg, NN, lane), rb1 = fetch_f16(vB1 + nbeg, NN, lane);
    commit_f16(lds[wave][0][0], ra0, lane);
    commit_f16(lds[wave][0][1], ra1, lane);
    commit_f16(lds[wave][0][2], rb0, lane);
    commit_f16(lds[wave][0][3], rb1, lane);

    int cur = 0;
    for (int n0 = nbeg; n0 < nend; n0 += 32) {
        lds_fence();
        v16h a0 = load_frag(lds[wave][cur][0], lane);
        v16h a1 = load_frag(lds[wave][cur][1], lane);
        v16h b0 = load_frag(lds[wave][cur][2], lane);
        v16h b1 = load_frag(lds[wave][cur][3], lane);
        const bool nxt = (n0 + 32) < nend;
        if (nxt) {
            ra0 = fetch_f16(kA0 + n0 + 32, NN, lane);
            ra1 = fetch_f16(kA1 + n0 + 32, NN, lane);
            rb0 = fetch_f16(vB0 + n0 + 32, NN, lane);
            rb1 = fetch_f16(vB1 + n0 + 32, NN, lane);
        }
        c00 = wmma16(a0, b0, c00);
        c01 = wmma16(a0, b1, c01);
        c10 = wmma16(a1, b0, c10);
        c11 = wmma16(a1, b1, c11);
        if (nxt) {
            commit_f16(lds[wave][cur ^ 1][0], ra0, lane);
            commit_f16(lds[wave][cur ^ 1][1], ra1, lane);
            commit_f16(lds[wave][cur ^ 1][2], rb0, lane);
            commit_f16(lds[wave][cur ^ 1][3], rb1, lane);
        }
        cur ^= 1;
    }

    float* awb = awp + (size_t)ks * BB * DD * DD + (size_t)b * DD * DD;
    const int ec = e0 + (lane & 15);
    const int dr = d0 + (lane >> 4) * 8;
#pragma unroll
    for (int v = 0; v < 8; ++v) {
        awb[(size_t)(dr + v) * DD + ec]           = c00[v];
        awb[(size_t)(dr + v) * DD + ec + 16]      = c01[v];
        awb[(size_t)(dr + 16 + v) * DD + ec]      = c10[v];
        awb[(size_t)(dr + 16 + v) * DD + ec + 16] = c11[v];
    }
}

// ===========================================================================
// Kernel 3: sum KSPLIT partials, softmax over d per (b,e) column (x 1/sqrt(B));
//           writes transposed f16: awT[b][e][d]
// ===========================================================================
__global__ __launch_bounds__(256) void ra_softmax_kernel(
    const float* __restrict__ awp, _Float16* __restrict__ awT) {
    const int lane = threadIdx.x & 31;
    const int wave = threadIdx.x >> 5;
    int col = blockIdx.x * SWPB + wave;              // 0..2047
    const int b = col >> 8;
    const int e = col & 255;

    const size_t part = (size_t)BB * DD * DD;
    const float* a = awp + (size_t)b * DD * DD + e;  // stride DD over d
    const float scale = 0.35355339059327373f;        // 1/sqrt(8)

    float x[8];
    float mx = -3.402823466e+38f;
#pragma unroll
    for (int j = 0; j < 8; ++j) {
        const size_t off = (size_t)(lane + 32 * j) * DD;
        float s = a[off];
#pragma unroll
        for (int p = 1; p < KSPLIT; ++p) s += a[p * part + off];
        x[j] = s * scale;
        mx = fmaxf(mx, x[j]);
    }
#pragma unroll
    for (int o = 16; o > 0; o >>= 1) mx = fmaxf(mx, __shfl_xor(mx, o, 32));

    float s = 0.0f;
#pragma unroll
    for (int j = 0; j < 8; ++j) { x[j] = __expf(x[j] - mx); s += x[j]; }
#pragma unroll
    for (int o = 16; o > 0; o >>= 1) s += __shfl_xor(s, o, 32);
    const float inv = 1.0f / s;

    _Float16* t = awT + (size_t)b * DD * DD + (size_t)e * DD;
#pragma unroll
    for (int j = 0; j < 8; ++j)
        t[lane + 32 * j] = (_Float16)(x[j] * inv);
}

// ===========================================================================
// Kernel 4: out[b,n,e] = sum_d q[b,n,d] * awT[b,e,d]   (f32 output)
// ===========================================================================
__global__ __launch_bounds__(128) void ra_out_kernel(
    const _Float16* __restrict__ qf, const _Float16* __restrict__ awT,
    float* __restrict__ out) {
    __shared__ __align__(16) _Float16 lds[WPB][2][4][16 * LDS_STRIDE];
    const int lane = threadIdx.x & 31;
    const int wave = threadIdx.x >> 5;

    // gt = (b * 256 + mb) * 8 + eb : 16384 waves
    int gt = blockIdx.x * WPB + wave;
    const int eb = gt & 7; int rem = gt >> 3;
    const int mb = rem & 255; rem >>= 8;
    const int b  = rem;

    const int row0 = mb * 32, col0 = eb * 32;
    const _Float16* qA0 = qf + ((size_t)b * NN + row0) * DD;
    const _Float16* qA1 = qf + ((size_t)b * NN + row0 + 16) * DD;
    const _Float16* wB0 = awT + ((size_t)b * DD + col0) * DD;
    const _Float16* wB1 = awT + ((size_t)b * DD + col0 + 16) * DD;

    v8f c00 = {}, c01 = {}, c10 = {}, c11 = {};

    F16Tile ra0 = fetch_f16(qA0, DD, lane), ra1 = fetch_f16(qA1, DD, lane);
    F16Tile rb0 = fetch_f16(wB0, DD, lane), rb1 = fetch_f16(wB1, DD, lane);
    commit_f16(lds[wave][0][0], ra0, lane);
    commit_f16(lds[wave][0][1], ra1, lane);
    commit_f16(lds[wave][0][2], rb0, lane);
    commit_f16(lds[wave][0][3], rb1, lane);

    int cur = 0;
#pragma unroll
    for (int kk = 0; kk < DD; kk += 32) {
        lds_fence();
        v16h a0 = load_frag(lds[wave][cur][0], lane);
        v16h a1 = load_frag(lds[wave][cur][1], lane);
        v16h b0 = load_frag(lds[wave][cur][2], lane);
        v16h b1 = load_frag(lds[wave][cur][3], lane);
        const bool nxt = (kk + 32) < DD;
        if (nxt) {
            ra0 = fetch_f16(qA0 + kk + 32, DD, lane);
            ra1 = fetch_f16(qA1 + kk + 32, DD, lane);
            rb0 = fetch_f16(wB0 + kk + 32, DD, lane);
            rb1 = fetch_f16(wB1 + kk + 32, DD, lane);
        }
        c00 = wmma16(a0, b0, c00);
        c01 = wmma16(a0, b1, c01);
        c10 = wmma16(a1, b0, c10);
        c11 = wmma16(a1, b1, c11);
        if (nxt) {
            commit_f16(lds[wave][cur ^ 1][0], ra0, lane);
            commit_f16(lds[wave][cur ^ 1][1], ra1, lane);
            commit_f16(lds[wave][cur ^ 1][2], rb0, lane);
            commit_f16(lds[wave][cur ^ 1][3], rb1, lane);
        }
        cur ^= 1;
    }

    float* ob = out + (size_t)b * NN * DD;
    const int e0 = col0 + (lane & 15);
    const int n0 = row0 + (lane >> 4) * 8;
#pragma unroll
    for (int v = 0; v < 8; ++v) {
        ob[(size_t)(n0 + v) * DD + e0]           = c00[v];
        ob[(size_t)(n0 + v) * DD + e0 + 16]      = c01[v];
        ob[(size_t)(n0 + 16 + v) * DD + e0]      = c10[v];
        ob[(size_t)(n0 + 16 + v) * DD + e0 + 16] = c11[v];
    }
}

// ===========================================================================
extern "C" void kernel_launch(void* const* d_in, const int* in_sizes, int n_in,
                              void* d_out, int out_size, void* d_ws, size_t ws_size,
                              hipStream_t stream) {
    const float* query = (const float*)d_in[0];
    const float* key   = (const float*)d_in[1];
    const float* value = (const float*)d_in[2];
    const unsigned char* mask = (const unsigned char*)d_in[3];  // bool, 1B/elem
    // d_in[4] = need_weights (ignored)
    const float* Wq = (const float*)d_in[5];
    const float* bq = (const float*)d_in[6];
    const float* Wk = (const float*)d_in[7];
    const float* bk = (const float*)d_in[8];
    const float* Wv = (const float*)d_in[9];
    const float* bv = (const float*)d_in[10];
    float* out = (float*)d_out;

    // workspace carve-up
    char* ws = (char*)d_ws;
    const size_t qkv_bytes = (size_t)BB * NN * DD * sizeof(_Float16);   // 32 MiB each
    const size_t aw_part   = (size_t)BB * DD * DD * sizeof(float);      // 2 MiB
    _Float16* qf  = (_Float16*)(ws);                                    // row-major
    _Float16* kT  = (_Float16*)(ws + qkv_bytes);                        // [b][d][n]
    _Float16* vT  = (_Float16*)(ws + 2 * qkv_bytes);                    // [b][e][n]
    float*    awp = (float*)   (ws + 3 * qkv_bytes);                    // 4 partials
    _Float16* awT = (_Float16*)(ws + 3 * qkv_bytes + KSPLIT * aw_part); // 1 MiB

    // 1) projections: 3 * 2048 * 8 waves / 4 per block
    ra_proj_kernel<<<dim3(3 * ((BB * NN) / 32) * (DD / 32) / WPB), dim3(128), 0, stream>>>(
        query, key, value, Wq, bq, Wk, bk, Wv, bv, mask, qf, kT, vT);

    // 2) aw partials = K^T V : 8*8*8*KSPLIT = 2048 waves
    ra_gram_kernel<<<dim3(BB * 8 * 8 * KSPLIT / WPB), dim3(128), 0, stream>>>(
        kT, vT, awp);

    // 3) sum partials + softmax over d, write transposed f16: 8*256 columns
    ra_softmax_kernel<<<dim3(BB * DD / SWPB), dim3(256), 0, stream>>>(awp, awT);

    // 4) out = Q @ aw : 8 * 256 * 8 waves
    ra_out_kernel<<<dim3(BB * (NN / 32) * (DD / 32) / WPB), dim3(128), 0, stream>>>(
        qf, awT, out);
}
